// VAE_Gumbel_8418135900358
// MI455X (gfx1250) — matrix operand order repaired and checked by reading, not compiled, not run on hardware
//
#include <hip/hip_runtime.h>

// ---------------- problem constants ----------------
constexpr int B  = 2048;
constexpr int D  = 2048;
constexpr int H  = 512;
constexpr int H2 = 1024;   // 2*H
constexpr int Z  = 500;
constexpr int ZP = 512;    // Z padded to multiple of 32
constexpr int KSEL = 64;
#define F_EPS   1e-10f
#define INV_T   10.0f

// ---------------- vector types ----------------
typedef __attribute__((ext_vector_type(8)))  unsigned short us8;
typedef __attribute__((ext_vector_type(16))) __bf16         v16bf;
typedef __attribute__((ext_vector_type(8)))  float          v8f;

union Frag16 { us8 u[2]; v16bf bf; };

__device__ __forceinline__ unsigned short f2bf(float f) {
  unsigned int u = __float_as_uint(f);
  unsigned int r = u + 0x7FFFu + ((u >> 16) & 1u);   // round-to-nearest-even
  return (unsigned short)(r >> 16);
}

// ---------------- weight prep: f32 [K,N] row-major -> bf16 [Npad,Kpad] col-major, zero-padded
__global__ __launch_bounds__(256) void prep_w(const float* __restrict__ W,
                                              unsigned short* __restrict__ Wt,
                                              int Kdim, int Ndim, int Kpad, int Npad) {
  int idx = blockIdx.x * 256 + threadIdx.x;
  if (idx >= Kpad * Npad) return;
  int n = idx / Kpad;
  int k = idx - n * Kpad;
  float v = (n < Ndim && k < Kdim) ? W[(size_t)k * Ndim + n] : 0.0f;
  Wt[idx] = f2bf(v);
}

// ---------------- f32 -> bf16 elementwise ----------------
__global__ __launch_bounds__(256) void cvt_bf16(const float* __restrict__ in,
                                                unsigned short* __restrict__ out, int n) {
  int i = blockIdx.x * 256 + threadIdx.x;
  if (i < n) out[i] = f2bf(in[i]);
}

// ---------------- WMMA bf16 GEMM:  out[M,N] = A[M,Kp] @ W + bias  (epi=1 -> sigmoid)
// A: bf16 row-major [M,Kp]; Wt: bf16 [Npad,Kp] (column-major of W). grid = (Npad/64, M/128)
// Kp must be a multiple of 64 (register double-buffered pipeline).
__global__ __launch_bounds__(256) void gemm_wmma_bf16(
    const unsigned short* __restrict__ A,
    const unsigned short* __restrict__ Wt,
    const float* __restrict__ bias,
    float* __restrict__ out,
    int Kp, int N, int epi) {
  const int tid  = threadIdx.x;
  const int wave = tid >> 5;
  const int lane = tid & 31;
  const int half = lane >> 4;
  const int lm   = lane & 15;
  const int m0   = blockIdx.y * 128 + wave * 16;
  const int n0   = blockIdx.x * 64;

  // A frag k order used by this code (matches B): half 0 -> K 0..15; half 1 -> K 16..31
  const unsigned short* aRow = A + (size_t)(m0 + lm) * Kp + half * 16;
  // B frag k order used by this code (matches A): lanes 0-15 K=0..15 of col, lanes 16-31 K=16..31
  const unsigned short* bp[4];
#pragma unroll
  for (int j = 0; j < 4; ++j)
    bp[j] = Wt + (size_t)(n0 + j * 16 + lm) * Kp + half * 16;

  v8f acc[4] = {};
  Frag16 a1, a2, b1[4], b2[4];

  // prologue: fragments for k=0
  a1.u[0] = *(const us8*)(aRow + 0);
  a1.u[1] = *(const us8*)(aRow + 8);
#pragma unroll
  for (int j = 0; j < 4; ++j) {
    b1[j].u[0] = *(const us8*)(bp[j] + 0);
    b1[j].u[1] = *(const us8*)(bp[j] + 8);
  }

  for (int k0 = 0; k0 < Kp; k0 += 64) {
    // prefetch k0+32 into set 2 while set 1 computes
    a2.u[0] = *(const us8*)(aRow + k0 + 32);
    a2.u[1] = *(const us8*)(aRow + k0 + 40);
#pragma unroll
    for (int j = 0; j < 4; ++j) {
      b2[j].u[0] = *(const us8*)(bp[j] + k0 + 32);
      b2[j].u[1] = *(const us8*)(bp[j] + k0 + 40);
    }
#pragma unroll
    for (int j = 0; j < 4; ++j)
      acc[j] = __builtin_amdgcn_wmma_f32_16x16x32_bf16(
          false, a1.bf, false, b1[j].bf, (short)0, acc[j], false, false);

    // prefetch k0+64 into set 1 while set 2 computes
    if (k0 + 64 < Kp) {
      a1.u[0] = *(const us8*)(aRow + k0 + 64);
      a1.u[1] = *(const us8*)(aRow + k0 + 72);
#pragma unroll
      for (int j = 0; j < 4; ++j) {
        b1[j].u[0] = *(const us8*)(bp[j] + k0 + 64);
        b1[j].u[1] = *(const us8*)(bp[j] + k0 + 72);
      }
    }
#pragma unroll
    for (int j = 0; j < 4; ++j)
      acc[j] = __builtin_amdgcn_wmma_f32_16x16x32_bf16(
          false, a2.bf, false, b2[j].bf, (short)0, acc[j], false, false);
  }

#pragma unroll
  for (int j = 0; j < 4; ++j) {
    int col = n0 + j * 16 + lm;
    if (col < N) {
      float bv = bias[col];
#pragma unroll
      for (int r = 0; r < 8; ++r) {
        float v = acc[j][r] + bv;                       // C/D layout: M = r + half*8
        if (epi == 1) v = 1.0f / (1.0f + __expf(-v));
        out[(size_t)(m0 + half * 8 + r) * N + col] = v;
      }
    }
  }
}

// ---------------- fused training-mode BatchNorm + activation + bf16 convert
// one thread per column; two coalesced passes over B rows (L2 resident).
__global__ __launch_bounds__(256) void bn_act(const float* __restrict__ Y,
                                              const float* __restrict__ g,
                                              const float* __restrict__ beta,
                                              unsigned short* __restrict__ out_bf,
                                              int N, float slope) {
  int n = blockIdx.x * 256 + threadIdx.x;
  if (n >= N) return;
  float s = 0.f, ss = 0.f;
  for (int r = 0; r < B; ++r) {
    float v = Y[(size_t)r * N + n];
    s += v; ss += v * v;
  }
  float mean = s * (1.0f / B);
  float var  = ss * (1.0f / B) - mean * mean;
  float rstd = rsqrtf(var + 1e-5f);
  float gg = g[n] * rstd;
  float bb = beta[n] - mean * gg;
  for (int r = 0; r < B; ++r) {
    float v = Y[(size_t)r * N + n] * gg + bb;
    v = (v >= 0.f) ? v : v * slope;                 // slope=0 -> ReLU, 0.01 -> LeakyReLU
    out_bf[(size_t)r * N + n] = f2bf(v);
  }
}

// ---------------- Gumbel subset scan: one block per row, 8 cols/thread in registers
__global__ __launch_bounds__(256) void gumbel_subset(
    const float* __restrict__ logits, const float* __restrict__ u,
    const float* __restrict__ x, unsigned short* __restrict__ xm) {
  __shared__ float redS[8];
  const int tid = threadIdx.x;
  const size_t base = (size_t)blockIdx.x * D;

  float w[8], kh[8], oh[8];
#pragma unroll
  for (int i = 0; i < 8; ++i) {
    int c = tid + i * 256;
    float uu = (1.0f - F_EPS) * u[base + c] + F_EPS;
    w[i]  = logits[base + c] + __logf(-__logf(uu));
    kh[i] = 0.f; oh[i] = 0.f;
  }

  for (int it = 0; it < KSEL; ++it) {
    float m = -3.4e38f;
#pragma unroll
    for (int i = 0; i < 8; ++i) {
      w[i] += __logf(fmaxf(1.0f - oh[i], F_EPS));   // log(1-0)=0 on first iter
      m = fmaxf(m, w[i]);
    }
    // wave32 shuffle max + cross-wave LDS combine
    for (int o = 16; o > 0; o >>= 1) m = fmaxf(m, __shfl_xor(m, o));
    if ((tid & 31) == 0) redS[tid >> 5] = m;
    __syncthreads();
    float M = redS[0];
#pragma unroll
    for (int q = 1; q < 8; ++q) M = fmaxf(M, redS[q]);
    __syncthreads();

    float s = 0.f;
#pragma unroll
    for (int i = 0; i < 8; ++i) { oh[i] = __expf((w[i] - M) * INV_T); s += oh[i]; }
    for (int o = 16; o > 0; o >>= 1) s += __shfl_xor(s, o);
    if ((tid & 31) == 0) redS[tid >> 5] = s;
    __syncthreads();
    float S = 0.f;
#pragma unroll
    for (int q = 0; q < 8; ++q) S += redS[q];
    __syncthreads();

    float invS = 1.0f / S;
#pragma unroll
    for (int i = 0; i < 8; ++i) { oh[i] *= invS; kh[i] += oh[i]; }
  }
#pragma unroll
  for (int i = 0; i < 8; ++i) {
    int c = tid + i * 256;
    xm[base + c] = f2bf(x[base + c] * kh[i]);
  }
}

// ---------------- reparameterization: z = mu + eps*exp(0.5*logvar), zero-padded to ZP cols
__global__ __launch_bounds__(256) void reparam(const float* __restrict__ mu,
                                               const float* __restrict__ logvar,
                                               const float* __restrict__ eps,
                                               unsigned short* __restrict__ z_bf) {
  int idx = blockIdx.x * 256 + threadIdx.x;       // over B*ZP
  if (idx >= B * ZP) return;
  int r = idx >> 9;          // / ZP
  int c = idx & (ZP - 1);
  float v = 0.f;
  if (c < Z) {
    size_t o = (size_t)r * Z + c;
    v = mu[o] + eps[o] * __expf(0.5f * logvar[o]);
  }
  z_bf[idx] = f2bf(v);
}

// =====================================================================
extern "C" void kernel_launch(void* const* d_in, const int* in_sizes, int n_in,
                              void* d_out, int out_size, void* d_ws, size_t ws_size,
                              hipStream_t stream) {
  (void)in_sizes; (void)n_in; (void)out_size; (void)ws_size;
  const float* x       = (const float*)d_in[0];
  const float* u_gum   = (const float*)d_in[1];
  const float* eps     = (const float*)d_in[2];
  const float* wc_w1   = (const float*)d_in[3];
  const float* wc_b1   = (const float*)d_in[4];
  const float* wc_g    = (const float*)d_in[5];
  const float* wc_beta = (const float*)d_in[6];
  const float* wc_w2   = (const float*)d_in[7];
  const float* wc_b2   = (const float*)d_in[8];
  const float* enc_w1  = (const float*)d_in[9];
  const float* enc_b1  = (const float*)d_in[10];
  const float* enc_g1  = (const float*)d_in[11];
  const float* enc_be1 = (const float*)d_in[12];
  const float* enc_w2  = (const float*)d_in[13];
  const float* enc_b2  = (const float*)d_in[14];
  const float* enc_g2  = (const float*)d_in[15];
  const float* enc_be2 = (const float*)d_in[16];
  const float* enc_w3  = (const float*)d_in[17];
  const float* enc_b3  = (const float*)d_in[18];
  const float* enc_g3  = (const float*)d_in[19];
  const float* enc_be3 = (const float*)d_in[20];
  const float* fc21_w  = (const float*)d_in[21];
  const float* fc21_b  = (const float*)d_in[22];
  const float* fc22_w  = (const float*)d_in[23];
  const float* fc22_b  = (const float*)d_in[24];
  const float* fc3_w   = (const float*)d_in[25];
  const float* fc3_b   = (const float*)d_in[26];
  const float* fc3_g   = (const float*)d_in[27];
  const float* fc3_be  = (const float*)d_in[28];
  const float* fc4_w   = (const float*)d_in[29];
  const float* fc4_b   = (const float*)d_in[30];

  float* mu_x   = (float*)d_out;                       // [B, D]
  float* mu     = (float*)d_out + (size_t)B * D;       // [B, Z]
  float* logvar = mu + (size_t)B * Z;                  // [B, Z]

  // ------- workspace carve-out -------
  size_t off = 0;
  char* base = (char*)d_ws;
  auto carve = [&](size_t bytes) -> void* {
    void* p = base + off;
    off += (bytes + 255) & ~size_t(255);
    return p;
  };
  unsigned short* wc_w1t = (unsigned short*)carve((size_t)H  * D  * 2);
  unsigned short* wc_w2t = (unsigned short*)carve((size_t)D  * H  * 2);
  unsigned short* e1t    = (unsigned short*)carve((size_t)H2 * D  * 2);
  unsigned short* e2t    = (unsigned short*)carve((size_t)H  * H2 * 2);
  unsigned short* e3t    = (unsigned short*)carve((size_t)H  * H  * 2);
  unsigned short* f21t   = (unsigned short*)carve((size_t)ZP * H  * 2);
  unsigned short* f22t   = (unsigned short*)carve((size_t)ZP * H  * 2);
  unsigned short* f3t    = (unsigned short*)carve((size_t)H  * ZP * 2);
  unsigned short* f4t    = (unsigned short*)carve((size_t)D  * H  * 2);
  unsigned short* x_bf   = (unsigned short*)carve((size_t)B * D  * 2);
  unsigned short* xm_bf  = (unsigned short*)carve((size_t)B * D  * 2);
  unsigned short* h_bf   = (unsigned short*)carve((size_t)B * H  * 2);
  unsigned short* h1_bf  = (unsigned short*)carve((size_t)B * H2 * 2);
  unsigned short* h2_bf  = (unsigned short*)carve((size_t)B * H  * 2);
  unsigned short* h3_bf  = (unsigned short*)carve((size_t)B * H  * 2);
  unsigned short* z_bf   = (unsigned short*)carve((size_t)B * ZP * 2);
  unsigned short* hd_bf  = (unsigned short*)carve((size_t)B * H  * 2);
  float*          Yf     = (float*)carve((size_t)B * D * 4);   // shared pre-BN f32 buffer

  auto pgrid = [](int elems) { return dim3((elems + 255) / 256); };

  // ------- weight prep (f32 [K,N] -> bf16 [Npad,Kpad] col-major) -------
  hipLaunchKernelGGL(prep_w, pgrid(H  * D ), dim3(256), 0, stream, wc_w1, wc_w1t, D,  H,  D,  H );
  hipLaunchKernelGGL(prep_w, pgrid(D  * H ), dim3(256), 0, stream, wc_w2, wc_w2t, H,  D,  H,  D );
  hipLaunchKernelGGL(prep_w, pgrid(H2 * D ), dim3(256), 0, stream, enc_w1, e1t,   D,  H2, D,  H2);
  hipLaunchKernelGGL(prep_w, pgrid(H  * H2), dim3(256), 0, stream, enc_w2, e2t,   H2, H,  H2, H );
  hipLaunchKernelGGL(prep_w, pgrid(H  * H ), dim3(256), 0, stream, enc_w3, e3t,   H,  H,  H,  H );
  hipLaunchKernelGGL(prep_w, pgrid(ZP * H ), dim3(256), 0, stream, fc21_w, f21t,  H,  Z,  H,  ZP);
  hipLaunchKernelGGL(prep_w, pgrid(ZP * H ), dim3(256), 0, stream, fc22_w, f22t,  H,  Z,  H,  ZP);
  hipLaunchKernelGGL(prep_w, pgrid(H  * ZP), dim3(256), 0, stream, fc3_w,  f3t,   Z,  H,  ZP, H );
  hipLaunchKernelGGL(prep_w, pgrid(D  * H ), dim3(256), 0, stream, fc4_w,  f4t,   H,  D,  H,  D );

  hipLaunchKernelGGL(cvt_bf16, pgrid(B * D), dim3(256), 0, stream, x, x_bf, B * D);

  // ------- weight_creator -------
  hipLaunchKernelGGL(gemm_wmma_bf16, dim3(H / 64, B / 128), dim3(256), 0, stream,
                     x_bf, wc_w1t, wc_b1, Yf, D, H, 0);                        // y1 = x@wc_w1+b
  hipLaunchKernelGGL(bn_act, dim3(H / 256), dim3(256), 0, stream,
                     Yf, wc_g, wc_beta, h_bf, H, 0.0f);                        // ReLU
  hipLaunchKernelGGL(gemm_wmma_bf16, dim3(D / 64, B / 128), dim3(256), 0, stream,
                     h_bf, wc_w2t, wc_b2, Yf, H, D, 0);                        // w_logits (in Yf)

  // ------- gumbel top-K scan + xm = x*khot -------
  hipLaunchKernelGGL(gumbel_subset, dim3(B), dim3(256), 0, stream, Yf, u_gum, x, xm_bf);

  // ------- encoder -------
  hipLaunchKernelGGL(gemm_wmma_bf16, dim3(H2 / 64, B / 128), dim3(256), 0, stream,
                     xm_bf, e1t, enc_b1, Yf, D, H2, 0);
  hipLaunchKernelGGL(bn_act, dim3(H2 / 256), dim3(256), 0, stream,
                     Yf, enc_g1, enc_be1, h1_bf, H2, 0.01f);
  hipLaunchKernelGGL(gemm_wmma_bf16, dim3(H / 64, B / 128), dim3(256), 0, stream,
                     h1_bf, e2t, enc_b2, Yf, H2, H, 0);
  hipLaunchKernelGGL(bn_act, dim3(H / 256), dim3(256), 0, stream,
                     Yf, enc_g2, enc_be2, h2_bf, H, 0.01f);
  hipLaunchKernelGGL(gemm_wmma_bf16, dim3(H / 64, B / 128), dim3(256), 0, stream,
                     h2_bf, e3t, enc_b3, Yf, H, H, 0);
  hipLaunchKernelGGL(bn_act, dim3(H / 256), dim3(256), 0, stream,
                     Yf, enc_g3, enc_be3, h3_bf, H, 0.01f);

  // ------- heads: mu / logvar straight into d_out -------
  hipLaunchKernelGGL(gemm_wmma_bf16, dim3(ZP / 64, B / 128), dim3(256), 0, stream,
                     h3_bf, f21t, fc21_b, mu, H, Z, 0);
  hipLaunchKernelGGL(gemm_wmma_bf16, dim3(ZP / 64, B / 128), dim3(256), 0, stream,
                     h3_bf, f22t, fc22_b, logvar, H, Z, 0);

  // ------- z = mu + eps*exp(0.5*logvar), padded to ZP -------
  hipLaunchKernelGGL(reparam, pgrid(B * ZP), dim3(256), 0, stream, mu, logvar, eps, z_bf);

  // ------- decoder -------
  hipLaunchKernelGGL(gemm_wmma_bf16, dim3(H / 64, B / 128), dim3(256), 0, stream,
                     z_bf, f3t, fc3_b, Yf, ZP, H, 0);
  hipLaunchKernelGGL(bn_act, dim3(H / 256), dim3(256), 0, stream,
                     Yf, fc3_g, fc3_be, hd_bf, H, 0.01f);
  hipLaunchKernelGGL(gemm_wmma_bf16, dim3(D / 64, B / 128), dim3(256), 0, stream,
                     hd_bf, f4t, fc4_b, mu_x, H, D, 1);                        // fused sigmoid
}
